// PointNetSeg_41661182771792
// MI455X (gfx1250) — matrix operand order repaired and checked by reading, not compiled, not run on hardware
//
#include <hip/hip_runtime.h>

// ---- CDNA5 WMMA types -------------------------------------------------------
typedef __attribute__((ext_vector_type(16))) __bf16 v16bf;
typedef __attribute__((ext_vector_type(8)))  float  v8f;
typedef __attribute__((ext_vector_type(4)))  float  f4;   // trivial 16B vector (avoid HIP float4 ctor in union)

#define NPTS   131072
#define NCLOUD 8
#define PPC    (NPTS / NCLOUD)

__device__ __forceinline__ unsigned short f2bf(float f) {
  unsigned u = __float_as_uint(f);
  u += 0x7FFFu + ((u >> 16) & 1u);           // round-to-nearest-even
  return (unsigned short)(u >> 16);
}

union Frag { v16bf v; f4 q[2]; };

__device__ __forceinline__ v8f wmma_bf16(const Frag& A, const Frag& B, v8f c) {
  return __builtin_amdgcn_wmma_f32_16x16x32_bf16(false, A.v, false, B.v,
                                                 (short)0, c, false, false);
}

// ---- generic WMMA layer: out = relu(in @ Wpacked + bias) --------------------
// Register-blocked over M: one wave owns an N-tile, loads each B fragment once
// per kt and applies it to all MT row-tiles (B traffic cut MT x).
// sin : LDS bf16, row-major [MT*16 x Kp], Kp multiple of 32, 16B aligned
// wp  : global bf16 packed B-fragments: ((nt*nK + kt)*32 + lane)*16 + j,
//       element j of lane (16*kh+n) holds W[kt*32 + 8*kh + (j&7) + 16*(j>>3)][nt*16 + n]
// sout: LDS bf16, row-major, stride Ns
template <int MT>
__device__ __forceinline__ void layer_ws(const unsigned short* sin, int Kp,
                                         unsigned short* sout, int Ns,
                                         const unsigned short* wp, const float* bias,
                                         int N) {
  const int lane = threadIdx.x & 31;
  const int wv   = threadIdx.x >> 5;   // 8 waves / block
  const int kh   = lane >> 4;
  const int ln   = lane & 15;
  const int nK   = Kp >> 5;
  const int nN   = N >> 4;
  for (int nt = wv; nt < nN; nt += 8) {
    v8f c[MT];
#pragma unroll
    for (int mt = 0; mt < MT; ++mt) c[mt] = (v8f){0.f,0.f,0.f,0.f,0.f,0.f,0.f,0.f};
    const unsigned short* wb = wp + (size_t)nt * nK * 512 + lane * 16;
    const unsigned short* ab = sin + ln * Kp + kh * 8;
    for (int kt = 0; kt < nK; ++kt) {
      Frag B;
      B.q[0] = *(const f4*)(wb + kt * 512);
      B.q[1] = *(const f4*)(wb + kt * 512 + 8);
#pragma unroll
      for (int mt = 0; mt < MT; ++mt) {
        Frag A;
        A.q[0] = *(const f4*)(ab + mt * 16 * Kp + kt * 32);
        A.q[1] = *(const f4*)(ab + mt * 16 * Kp + kt * 32 + 16);
        c[mt] = wmma_bf16(A, B, c[mt]);
      }
    }
    const float bb = bias[nt * 16 + ln];
#pragma unroll
    for (int mt = 0; mt < MT; ++mt)
#pragma unroll
      for (int i = 0; i < 8; ++i) {
        float vv = fmaxf(c[mt][i] + bb, 0.f);
        sout[(mt * 16 + i + kh * 8) * Ns + nt * 16 + ln] = f2bf(vv);
      }
  }
}

// ---- WMMA layer fused with column-max reduce (relu implicit: max starts 0) --
template <int MT>
__device__ __forceinline__ void layer_max(const unsigned short* sin, int Kp,
                                          const unsigned short* wp, const float* bias,
                                          int N, unsigned* sred) {
  const int lane = threadIdx.x & 31;
  const int wv   = threadIdx.x >> 5;
  const int kh   = lane >> 4;
  const int ln   = lane & 15;
  const int nK   = Kp >> 5;
  const int nN   = N >> 4;
  for (int nt = wv; nt < nN; nt += 8) {
    v8f c[MT];
#pragma unroll
    for (int mt = 0; mt < MT; ++mt) c[mt] = (v8f){0.f,0.f,0.f,0.f,0.f,0.f,0.f,0.f};
    const unsigned short* wb = wp + (size_t)nt * nK * 512 + lane * 16;
    const unsigned short* ab = sin + ln * Kp + kh * 8;
    for (int kt = 0; kt < nK; ++kt) {
      Frag B;
      B.q[0] = *(const f4*)(wb + kt * 512);
      B.q[1] = *(const f4*)(wb + kt * 512 + 8);
#pragma unroll
      for (int mt = 0; mt < MT; ++mt) {
        Frag A;
        A.q[0] = *(const f4*)(ab + mt * 16 * Kp + kt * 32);
        A.q[1] = *(const f4*)(ab + mt * 16 * Kp + kt * 32 + 16);
        c[mt] = wmma_bf16(A, B, c[mt]);
      }
    }
    const float bb = bias[nt * 16 + ln];
    float m = 0.f;                       // relu folded into max-with-0
#pragma unroll
    for (int mt = 0; mt < MT; ++mt)
#pragma unroll
      for (int i = 0; i < 8; ++i) m = fmaxf(m, c[mt][i] + bb);
    m = fmaxf(m, __shfl_xor(m, 16, 32)); // merge M halves (same column)
    if (lane < 16) atomicMax(&sred[nt * 16 + ln], __float_as_uint(m));
  }
}

// ---- fused per-point MLP stack (Kin -> 64 -> 128 -> 1024) + segment max ----
__global__ void __launch_bounds__(256)
k_local_stack(const float* xin, int Kin, int Kp0,
              const unsigned short* w0, const float* b0,
              const unsigned short* w1, const float* b1,
              const unsigned short* w2, const float* b2,
              unsigned* g /* [8,1024] float bits */) {
  __shared__ alignas(16) unsigned short sIn[64 * 64];
  __shared__ alignas(16) unsigned short sH1[64 * 64];
  __shared__ alignas(16) unsigned short sH2[64 * 128];
  __shared__ unsigned sred[1024];
  const int tid = threadIdx.x;
  const int p0  = blockIdx.x * 64;      // 64 | 16384: block never straddles clouds
  const int cloud = p0 / PPC;
  for (int i = tid; i < 64 * Kp0; i += 256) {
    int r = i / Kp0, cc = i % Kp0;
    float v = (cc < Kin) ? xin[(size_t)(p0 + r) * Kin + cc] : 0.f;
    sIn[i] = f2bf(v);
  }
  for (int i = tid; i < 1024; i += 256) sred[i] = 0u;
  __syncthreads();
  layer_ws<4>(sIn, Kp0, sH1, 64, w0, b0, 64);
  __syncthreads();
  layer_ws<4>(sH1, 64, sH2, 128, w1, b1, 128);
  __syncthreads();
  layer_max<4>(sH2, 128, w2, b2, 1024, sred);
  __syncthreads();
  for (int i = tid; i < 1024; i += 256) atomicMax(&g[cloud * 1024 + i], sred[i]);
}

// ---- seg head: xf(64) -> 512 (per-cloud bias cb) -> 256 -> 128 -> 4 --------
__global__ void __launch_bounds__(256)
k_seg(const float* xf,
      const unsigned short* w0, const float* cb,      // 64->512, bias cb[cloud]
      const unsigned short* w1, const float* b1,      // 512->256
      const unsigned short* w2, const float* b2,      // 256->128
      const unsigned short* w3, const float* b3,      // 128->4 (packed to N=16)
      float* out) {
  __shared__ alignas(16) unsigned short sIn[32 * 64];
  __shared__ alignas(16) unsigned short sH1[32 * 512];
  __shared__ alignas(16) unsigned short sH2[32 * 256];
  __shared__ alignas(16) unsigned short sH3[32 * 128];
  const int tid = threadIdx.x;
  const int p0  = blockIdx.x * 32;
  const int cloud = p0 / PPC;
  for (int i = tid; i < 32 * 64; i += 256) {
    int r = i >> 6, cc = i & 63;
    sIn[i] = f2bf(xf[(size_t)(p0 + r) * 64 + cc]);
  }
  __syncthreads();
  layer_ws<2>(sIn, 64, sH1, 512, w0, cb + cloud * 512, 512);
  __syncthreads();
  layer_ws<2>(sH1, 512, sH2, 256, w1, b1, 256);
  __syncthreads();
  layer_ws<2>(sH2, 256, sH3, 128, w2, b2, 128);
  __syncthreads();
  const int lane = threadIdx.x & 31;
  const int wv = threadIdx.x >> 5;
  const int kh = lane >> 4, ln = lane & 15;
  if (wv == 0) {                            // final 128->4: one wave, both M tiles
    v8f c[2];
    c[0] = (v8f){0.f,0.f,0.f,0.f,0.f,0.f,0.f,0.f};
    c[1] = c[0];
    const unsigned short* ab = sH3 + ln * 128 + kh * 8;
    for (int kt = 0; kt < 4; ++kt) {
      Frag B;
      const unsigned short* wl = w3 + (kt * 32 + lane) * 16;
      B.q[0] = *(const f4*)(wl);
      B.q[1] = *(const f4*)(wl + 8);
#pragma unroll
      for (int mt = 0; mt < 2; ++mt) {
        Frag A;
        A.q[0] = *(const f4*)(ab + mt * 16 * 128 + kt * 32);
        A.q[1] = *(const f4*)(ab + mt * 16 * 128 + kt * 32 + 16);
        c[mt] = wmma_bf16(A, B, c[mt]);
      }
    }
    if (ln < 4) {
      const float bb = b3[ln];
#pragma unroll
      for (int mt = 0; mt < 2; ++mt)
#pragma unroll
        for (int i = 0; i < 8; ++i)
          out[(size_t)(p0 + mt * 16 + i + kh * 8) * 4 + ln] = fmaxf(c[mt][i] + bb, 0.f);
    }
  }
}

// ---- pack fp32 weights [K x N] (row-major, leading dim ld) into B-fragments -
__global__ void k_pack(const float* W, int rowOff, int K, int ld, int N,
                       unsigned short* dst, int total) {
  int idx = blockIdx.x * 256 + threadIdx.x;
  if (idx >= total) return;
  int j    = idx & 15;
  int lane = (idx >> 4) & 31;
  int tkt  = idx >> 9;
  int nK   = (K + 31) >> 5;
  int nt = tkt / nK, kt = tkt % nK;
  int krow = kt * 32 + ((lane >> 4) * 8) + (j & 7) + ((j >> 3) * 16);
  int ncol = nt * 16 + (lane & 15);
  float v = (krow < K && ncol < N) ? W[(size_t)(rowOff + krow) * ld + ncol] : 0.f;
  dst[idx] = f2bf(v);
}

__global__ void k_zero(unsigned* p, int n) {
  int i = blockIdx.x * 256 + threadIdx.x;
  if (i < n) p[i] = 0u;
}

// ---- tiny per-cloud global MLP: [8,1024]->512->256->fc(F) + I_k, scalar f32 -
__global__ void __launch_bounds__(256)
k_glob(const float* g, const float* W1, const float* b1,
       const float* W2, const float* b2,
       const float* Wfc, const float* bfc, float* trans, int F, int k) {
  __shared__ float sg[8 * 1024];
  __shared__ float sh1[8 * 512];
  __shared__ float sh2[8 * 256];
  const int tid = threadIdx.x;
  for (int i = tid; i < 8192; i += 256) sg[i] = g[i];
  __syncthreads();
  for (int o = tid; o < 8 * 512; o += 256) {
    int b = o >> 9, n = o & 511;
    float acc = b1[n];
    for (int kk = 0; kk < 1024; ++kk) acc += sg[b * 1024 + kk] * W1[kk * 512 + n];
    sh1[o] = fmaxf(acc, 0.f);
  }
  __syncthreads();
  for (int o = tid; o < 8 * 256; o += 256) {
    int b = o >> 8, n = o & 255;
    float acc = b2[n];
    for (int kk = 0; kk < 512; ++kk) acc += sh1[b * 512 + kk] * W2[kk * 256 + n];
    sh2[o] = fmaxf(acc, 0.f);
  }
  __syncthreads();
  for (int o = tid; o < 8 * F; o += 256) {
    int b = o / F, n = o % F;
    float acc = bfc[n];
    for (int kk = 0; kk < 256; ++kk) acc += sh2[b * 256 + kk] * Wfc[kk * F + n];
    if (n / k == n % k) acc += 1.f;                      // + identity
    trans[o] = acc;
  }
}

// ---- cb[b] = g3[b] @ Wseg0[64:1088] + b0   (per-cloud seg-layer1 bias) -----
__global__ void __launch_bounds__(256)
k_segbias(const float* g3, const float* W, const float* b, float* cb) {
  __shared__ float sg[8 * 1024];
  const int tid = threadIdx.x;
  for (int i = tid; i < 8192; i += 256) sg[i] = g3[i];
  __syncthreads();
  for (int o = tid; o < 8 * 512; o += 256) {
    int bc = o >> 9, n = o & 511;
    float acc = b[n];
    for (int kk = 0; kk < 1024; ++kk) acc += sg[bc * 1024 + kk] * W[(size_t)(64 + kk) * 512 + n];
    cb[o] = acc;
  }
}

// ---- x1 = relu-MLP(local1, x @ T3[cloud]) ----------------------------------
__global__ void __launch_bounds__(256)
k_apply1(const float* x, const float* trans,
         const float* W1, const float* b1, const float* W2, const float* b2,
         float* x1o) {
  __shared__ float sW2[64 * 64];
  __shared__ float sW1[3 * 64];
  __shared__ float sb1[64], sb2[64], sT[9];
  const int tid = threadIdx.x;
  const int p = blockIdx.x * 256 + tid;
  const int cloud = (blockIdx.x * 256) / PPC;
  for (int i = tid; i < 4096; i += 256) sW2[i] = W2[i];
  if (tid < 192) sW1[tid] = W1[tid];
  if (tid < 64) { sb1[tid] = b1[tid]; sb2[tid] = b2[tid]; }
  if (tid < 9)  sT[tid] = trans[cloud * 9 + tid];
  __syncthreads();
  float x0 = x[(size_t)p * 3], xx1 = x[(size_t)p * 3 + 1], x2 = x[(size_t)p * 3 + 2];
  float y0 = x0 * sT[0] + xx1 * sT[3] + x2 * sT[6];
  float y1 = x0 * sT[1] + xx1 * sT[4] + x2 * sT[7];
  float y2 = x0 * sT[2] + xx1 * sT[5] + x2 * sT[8];
  float h[64];
#pragma unroll
  for (int j = 0; j < 64; ++j)
    h[j] = fmaxf(y0 * sW1[j] + y1 * sW1[64 + j] + y2 * sW1[128 + j] + sb1[j], 0.f);
  for (int j = 0; j < 64; ++j) {
    float acc = sb2[j];
#pragma unroll
    for (int i = 0; i < 64; ++i) acc += h[i] * sW2[i * 64 + j];
    x1o[(size_t)p * 64 + j] = fmaxf(acc, 0.f);
  }
}

// ---- xf = x1 @ T64[cloud] ---------------------------------------------------
__global__ void __launch_bounds__(256)
k_apply2(const float* x1, const float* trans, float* xf) {
  __shared__ float sT[4096];
  const int tid = threadIdx.x;
  const int p = blockIdx.x * 256 + tid;
  const int cloud = (blockIdx.x * 256) / PPC;
  for (int i = tid; i < 4096; i += 256) sT[i] = trans[cloud * 4096 + i];
  __syncthreads();
  float v[64];
#pragma unroll
  for (int i = 0; i < 64; ++i) v[i] = x1[(size_t)p * 64 + i];
  for (int j = 0; j < 64; ++j) {
    float acc = 0.f;
#pragma unroll
    for (int i = 0; i < 64; ++i) acc += v[i] * sT[i * 64 + j];
    xf[(size_t)p * 64 + j] = acc;
  }
}

// ---- host orchestration -----------------------------------------------------
extern "C" void kernel_launch(void* const* d_in, const int* in_sizes, int n_in,
                              void* d_out, int out_size, void* d_ws, size_t ws_size,
                              hipStream_t stream) {
  (void)in_sizes; (void)n_in; (void)out_size; (void)ws_size;
  const float* x = (const float*)d_in[0];
  // param pointers per setup_inputs() flatten order (each layer = W,b)
  const float* s1l_w0 = (const float*)d_in[2];  const float* s1l_b0 = (const float*)d_in[3];
  const float* s1l_w1 = (const float*)d_in[4];  const float* s1l_b1 = (const float*)d_in[5];
  const float* s1l_w2 = (const float*)d_in[6];  const float* s1l_b2 = (const float*)d_in[7];
  const float* s1g_w0 = (const float*)d_in[8];  const float* s1g_b0 = (const float*)d_in[9];
  const float* s1g_w1 = (const float*)d_in[10]; const float* s1g_b1 = (const float*)d_in[11];
  const float* s1f_w  = (const float*)d_in[12]; const float* s1f_b  = (const float*)d_in[13];
  const float* l1_w0  = (const float*)d_in[14]; const float* l1_b0  = (const float*)d_in[15];
  const float* l1_w1  = (const float*)d_in[16]; const float* l1_b1  = (const float*)d_in[17];
  const float* s2l_w0 = (const float*)d_in[18]; const float* s2l_b0 = (const float*)d_in[19];
  const float* s2l_w1 = (const float*)d_in[20]; const float* s2l_b1 = (const float*)d_in[21];
  const float* s2l_w2 = (const float*)d_in[22]; const float* s2l_b2 = (const float*)d_in[23];
  const float* s2g_w0 = (const float*)d_in[24]; const float* s2g_b0 = (const float*)d_in[25];
  const float* s2g_w1 = (const float*)d_in[26]; const float* s2g_b1 = (const float*)d_in[27];
  const float* s2f_w  = (const float*)d_in[28]; const float* s2f_b  = (const float*)d_in[29];
  const float* l2_w0  = (const float*)d_in[30]; const float* l2_b0  = (const float*)d_in[31];
  const float* l2_w1  = (const float*)d_in[32]; const float* l2_b1  = (const float*)d_in[33];
  const float* l2_w2  = (const float*)d_in[34]; const float* l2_b2  = (const float*)d_in[35];
  const float* sg_w0  = (const float*)d_in[36]; const float* sg_b0  = (const float*)d_in[37];
  const float* sg_w1  = (const float*)d_in[38]; const float* sg_b1  = (const float*)d_in[39];
  const float* sg_w2  = (const float*)d_in[40]; const float* sg_b2  = (const float*)d_in[41];
  const float* sg_w3  = (const float*)d_in[42]; const float* sg_b3  = (const float*)d_in[43];

  char* base = (char*)d_ws; size_t off = 0;
  auto alloc = [&](size_t bytes) { void* p = base + off; off = (off + bytes + 255) & ~(size_t)255; return p; };
  unsigned* g1 = (unsigned*)alloc(8192 * 4);
  unsigned* g2 = (unsigned*)alloc(8192 * 4);
  unsigned* g3 = (unsigned*)alloc(8192 * 4);
  float* trans1 = (float*)alloc(72 * 4);
  float* trans2 = (float*)alloc(32768 * 4);
  float* cb = (float*)alloc(4096 * 4);
  float* x1 = (float*)alloc((size_t)NPTS * 64 * 4);
  float* xf = (float*)alloc((size_t)NPTS * 64 * 4);
  auto psz = [](int K, int N) { return (size_t)(((K + 31) & ~31) * ((N + 15) & ~15)); };
  unsigned short* pw[13];
  const int PK[13] = {3, 64, 128, 64, 64, 128, 64, 64, 128, 64, 512, 256, 128};
  const int PN[13] = {64, 128, 1024, 64, 128, 1024, 64, 128, 1024, 512, 256, 128, 4};
  for (int i = 0; i < 13; ++i) pw[i] = (unsigned short*)alloc(psz(PK[i], PN[i]) * 2);

  k_zero<<<32, 256, 0, stream>>>(g1, 8192);
  k_zero<<<32, 256, 0, stream>>>(g2, 8192);
  k_zero<<<32, 256, 0, stream>>>(g3, 8192);

  auto pack = [&](const float* W, int rowOff, int K, int ld, int N, unsigned short* dst) {
    int total = (int)psz(K, N);
    k_pack<<<(total + 255) / 256, 256, 0, stream>>>(W, rowOff, K, ld, N, dst, total);
  };
  pack(s1l_w0, 0, 3, 64, 64, pw[0]);      pack(s1l_w1, 0, 64, 128, 128, pw[1]);
  pack(s1l_w2, 0, 128, 1024, 1024, pw[2]);
  pack(s2l_w0, 0, 64, 64, 64, pw[3]);     pack(s2l_w1, 0, 64, 128, 128, pw[4]);
  pack(s2l_w2, 0, 128, 1024, 1024, pw[5]);
  pack(l2_w0, 0, 64, 64, 64, pw[6]);      pack(l2_w1, 0, 64, 128, 128, pw[7]);
  pack(l2_w2, 0, 128, 1024, 1024, pw[8]);
  pack(sg_w0, 0, 64, 512, 512, pw[9]);    // xf-rows of seg layer 1
  pack(sg_w1, 0, 512, 256, 256, pw[10]);
  pack(sg_w2, 0, 256, 128, 128, pw[11]);
  pack(sg_w3, 0, 128, 4, 4, pw[12]);

  // STN1 -> g1 -> trans1 ; x1 = local1(x @ T3)
  k_local_stack<<<NPTS / 64, 256, 0, stream>>>(x, 3, 32, pw[0], s1l_b0, pw[1], s1l_b1, pw[2], s1l_b2, g1);
  k_glob<<<1, 256, 0, stream>>>((const float*)g1, s1g_w0, s1g_b0, s1g_w1, s1g_b1, s1f_w, s1f_b, trans1, 9, 3);
  k_apply1<<<NPTS / 256, 256, 0, stream>>>(x, trans1, l1_w0, l1_b0, l1_w1, l1_b1, x1);
  // STN2 -> g2 -> trans2 ; xf = x1 @ T64
  k_local_stack<<<NPTS / 64, 256, 0, stream>>>(x1, 64, 64, pw[3], s2l_b0, pw[4], s2l_b1, pw[5], s2l_b2, g2);
  k_glob<<<1, 256, 0, stream>>>((const float*)g2, s2g_w0, s2g_b0, s2g_w1, s2g_b1, s2f_w, s2f_b, trans2, 4096, 64);
  k_apply2<<<NPTS / 256, 256, 0, stream>>>(x1, trans2, xf);
  // local2 -> g3 ; cb = g3 @ Wseg0[64:] + b ; seg head
  k_local_stack<<<NPTS / 64, 256, 0, stream>>>(xf, 64, 64, pw[6], l2_b0, pw[7], l2_b1, pw[8], l2_b2, g3);
  k_segbias<<<1, 256, 0, stream>>>((const float*)g3, sg_w0, sg_b0, cb);
  k_seg<<<NPTS / 32, 256, 0, stream>>>(xf, pw[9], cb, pw[10], sg_b1, pw[11], sg_b2, pw[12], sg_b3, (float*)d_out);
}
